// Model_61357902790974
// MI455X (gfx1250) — matrix operand order repaired and checked by reading, not compiled, not run on hardware
//
#include <hip/hip_runtime.h>
#include <hip/hip_bf16.h>
#include <stdint.h>

// ---------------------------------------------------------------------------
// BiLSTM-CRF for MI455X (gfx1250, wave32, WMMA bf16 -> f32 accum)
// ---------------------------------------------------------------------------
typedef __attribute__((ext_vector_type(16))) __bf16 v16bf;
typedef __attribute__((ext_vector_type(8)))  __bf16 v8bf;
typedef __attribute__((ext_vector_type(8)))  float  v8f;

#define TB      32768          // T*B = 512*64 rows
#define G4H     1200           // 4*H gate width
#define KH      320            // padded hidden K (300 -> 320)
#define K2H     640            // padded 2H K (600 -> 640)
#define HSTRIDE 328            // LDS h row stride (bf16), 16B-aligned rows
#define GSTRIDE 1208           // LDS gate row stride (f32)
#define NEGINF  -10000.0f

__device__ __forceinline__ float sigm(float x) { return 1.0f / (1.0f + __expf(-x)); }

// ---------------- embedding gather + f32->bf16 + K-pad -----------------------
__global__ void __launch_bounds__(256) embed_kernel(const int* __restrict__ x,
                                                    const float* __restrict__ emb,
                                                    __bf16* __restrict__ X0) {
    size_t i = (size_t)blockIdx.x * 256 + threadIdx.x;      // over 32768*80
    if (i >= (size_t)TB * 80) return;
    int r  = (int)(i / 80);            // r = t*64 + b
    int k4 = (int)(i % 80) * 4;
    int t = r >> 6, b = r & 63;
    __bf16* dst = X0 + (size_t)r * KH + k4;
    if (k4 < 300) {
        int tok = x[b * 512 + t];
        const float* e = emb + (size_t)tok * 300 + k4;
        dst[0] = (__bf16)e[0]; dst[1] = (__bf16)e[1];
        dst[2] = (__bf16)e[2]; dst[3] = (__bf16)e[3];
    } else {
        dst[0] = (__bf16)0.f; dst[1] = (__bf16)0.f;
        dst[2] = (__bf16)0.f; dst[3] = (__bf16)0.f;
    }
}

// ---------------- f32 -> bf16 convert with zero-pad (+hf/hb split remap) -----
__global__ void __launch_bounds__(256) cvt_pad_kernel(const float* __restrict__ src,
                                                      __bf16* __restrict__ dst,
                                                      int srcRows, int srcCols,
                                                      int dstCols, int split, int total) {
    int i = blockIdx.x * 256 + threadIdx.x;
    if (i >= total) return;
    int r = i / dstCols, c = i % dstCols;
    float v = 0.f;
    if (r < srcRows) {
        int sc = -1;
        if (!split) { if (c < srcCols) sc = c; }
        else        { int h = c / 320, cc = c % 320; if (cc < 300) sc = h * 300 + cc; }
        if (sc >= 0) v = src[(size_t)r * srcCols + sc];
    }
    dst[i] = (__bf16)v;
}

__global__ void __launch_bounds__(256) bias_sum_kernel(const float* __restrict__ bi,
                                                       const float* __restrict__ bh,
                                                       float* __restrict__ o, int n) {
    int i = blockIdx.x * 256 + threadIdx.x;
    if (i < n) o[i] = bi[i] + bh[i];
}

__global__ void __launch_bounds__(256) zero_bf16_kernel(__bf16* __restrict__ p, size_t n) {
    size_t i = (size_t)blockIdx.x * 256 + threadIdx.x;
    if (i < n) p[i] = (__bf16)0.f;
}

// ---------------- input projection: pre = X·Wi^T + (bi+bh), all (t,b) --------
// grid: (19200, 2)  block: 256 (8 waves); one 16x16 tile per wave.
__global__ void __launch_bounds__(256) gemm_pre_kernel(const __bf16* __restrict__ X,
                                                       const __bf16* __restrict__ WiF,
                                                       const __bf16* __restrict__ WiB,
                                                       const float* __restrict__ biasF,
                                                       const float* __restrict__ biasB,
                                                       __bf16* __restrict__ preF,
                                                       __bf16* __restrict__ preB,
                                                       int Kpad) {
    const int dir = blockIdx.y;
    const __bf16* W    = dir ? WiB   : WiF;
    const float*  bias = dir ? biasB : biasF;
    __bf16*       pre  = dir ? preB  : preF;

    int wave = threadIdx.x >> 5, lane = threadIdx.x & 31;
    int tile = blockIdx.x * 8 + wave;          // 0 .. 2048*75-1
    int mt = tile / 75, nt = tile % 75;
    int rbase = mt * 16;
    int m = lane & 15, half = lane >> 4;
    int s = half * 8, kh = half * 8;
    int n = nt * 16 + (lane & 15);

    v8f acc = {};
    for (int k0 = 0; k0 < Kpad; k0 += 32) {
        const __bf16* ap = X + (size_t)(rbase + m) * Kpad + k0 + s;
        v8bf a0 = *(const v8bf*)ap;
        v8bf a1 = *(const v8bf*)(ap + 16);
        v16bf a;
#pragma unroll
        for (int e = 0; e < 8; ++e) { a[e] = a0[e]; a[e + 8] = a1[e]; }
        const __bf16* bp = W + (size_t)n * Kpad + k0 + kh;
        v8bf b0 = *(const v8bf*)bp;
        v8bf b1 = *(const v8bf*)(bp + 16);
        v16bf b;
#pragma unroll
        for (int e = 0; e < 8; ++e) { b[e] = b0[e]; b[e + 8] = b1[e]; }
        acc = __builtin_amdgcn_wmma_f32_16x16x32_bf16(false, a, false, b,
                                                      (short)0, acc, false, false);
    }
    float bv = bias[n];
    int r0 = rbase + half * 8;
#pragma unroll
    for (int v = 0; v < 8; ++v)
        pre[(size_t)(r0 + v) * G4H + n] = (__bf16)(acc[v] + bv);
}

// ---------------- persistent recurrent scan: one block per direction ---------
// block: 1024 threads (32 waves); dynamic LDS = h(64x328 bf16) + gates(32x1208 f32)
__global__ void __launch_bounds__(1024) lstm_scan_kernel(const __bf16* __restrict__ WhF,
                                                         const __bf16* __restrict__ WhB,
                                                         const __bf16* __restrict__ preF,
                                                         const __bf16* __restrict__ preB,
                                                         __bf16* __restrict__ hout) {
    extern __shared__ char smem[];
    __bf16* h_lds = (__bf16*)smem;                                   // [64][HSTRIDE]
    float*  gates = (float*)(smem + 64 * HSTRIDE * sizeof(__bf16));  // [32][GSTRIDE]

    const int dir = blockIdx.x;
    const __bf16* Wh  = dir ? WhB  : WhF;
    const __bf16* pre = dir ? preB : preF;
    const int colOff  = dir * 320;

    const int tid  = threadIdx.x;
    const int wave = tid >> 5, lane = tid & 31;
    const int m = lane & 15, half = lane >> 4;
    const int s = half * 8, kh = half * 8;

    for (int i = tid; i < 64 * HSTRIDE; i += 1024) h_lds[i] = (__bf16)0.f;
    float cst[2][10];
#pragma unroll
    for (int p = 0; p < 2; ++p)
#pragma unroll
        for (int k = 0; k < 10; ++k) cst[p][k] = 0.f;
    __syncthreads();

    for (int step = 0; step < 512; ++step) {
        const int t = dir ? (511 - step) : step;
        const __bf16* pre_t = pre + (size_t)t * 64 * G4H;
        for (int pass = 0; pass < 2; ++pass) {
            // ---- gates[0:32][0:1200] = h[pass rows] · Wh^T (WMMA, f32 accum) ----
            for (int tt = wave; tt < 150; tt += 32) {
                int mt = tt / 75, nt = tt % 75;
                int row = pass * 32 + mt * 16 + m;
                int n = nt * 16 + (lane & 15);
                v8f acc = {};
                for (int k0 = 0; k0 < KH; k0 += 32) {
                    const __bf16* ap = h_lds + row * HSTRIDE + k0 + s;
                    v8bf a0 = *(const v8bf*)ap;
                    v8bf a1 = *(const v8bf*)(ap + 16);
                    v16bf a;
#pragma unroll
                    for (int e = 0; e < 8; ++e) { a[e] = a0[e]; a[e + 8] = a1[e]; }
                    const __bf16* bp = Wh + (size_t)n * KH + k0 + kh;
                    v8bf b0 = *(const v8bf*)bp;
                    v8bf b1 = *(const v8bf*)(bp + 16);
                    v16bf b;
#pragma unroll
                    for (int e = 0; e < 8; ++e) { b[e] = b0[e]; b[e + 8] = b1[e]; }
                    acc = __builtin_amdgcn_wmma_f32_16x16x32_bf16(false, a, false, b,
                                                                  (short)0, acc, false, false);
                }
                int r0 = mt * 16 + half * 8;
#pragma unroll
                for (int v = 0; v < 8; ++v) gates[(r0 + v) * GSTRIDE + n] = acc[v];
            }
            __syncthreads();
            // ---- elementwise LSTM cell update; c lives in registers --------------
#pragma unroll
            for (int k = 0; k < 10; ++k) {
                int idx = tid + k * 1024;
                if (idx < 9600) {
                    int r = idx / 300, j = idx - r * 300;
                    int b = pass * 32 + r;
                    float gi = gates[r * GSTRIDE + j]       + (float)pre_t[b * G4H + j];
                    float gf = gates[r * GSTRIDE + 300 + j] + (float)pre_t[b * G4H + 300 + j];
                    float gg = gates[r * GSTRIDE + 600 + j] + (float)pre_t[b * G4H + 600 + j];
                    float go = gates[r * GSTRIDE + 900 + j] + (float)pre_t[b * G4H + 900 + j];
                    float cn = sigm(gf) * cst[pass][k] + sigm(gi) * tanhf(gg);
                    cst[pass][k] = cn;
                    float hn = sigm(go) * tanhf(cn);
                    h_lds[b * HSTRIDE + j] = (__bf16)hn;
                    hout[((size_t)t * 64 + b) * K2H + colOff + j] = (__bf16)hn;
                }
            }
            __syncthreads();
        }
    }
}

// ---------------- output projection: feats = H2·Wout^T + bout ----------------
__global__ void __launch_bounds__(256) gemm_feats_kernel(const __bf16* __restrict__ H2,
                                                         const __bf16* __restrict__ Woutbf,
                                                         const float* __restrict__ bout,
                                                         float* __restrict__ feats) {
    int wave = threadIdx.x >> 5, lane = threadIdx.x & 31;
    int mt = blockIdx.x * 8 + wave;            // 0..2047
    int rbase = mt * 16;
    int m = lane & 15, half = lane >> 4;
    int s = half * 8, kh = half * 8;
    int n = lane & 15;

    v8f acc = {};
    for (int k0 = 0; k0 < K2H; k0 += 32) {
        const __bf16* ap = H2 + (size_t)(rbase + m) * K2H + k0 + s;
        v8bf a0 = *(const v8bf*)ap;
        v8bf a1 = *(const v8bf*)(ap + 16);
        v16bf a;
#pragma unroll
        for (int e = 0; e < 8; ++e) { a[e] = a0[e]; a[e + 8] = a1[e]; }
        const __bf16* bp = Woutbf + (size_t)n * K2H + k0 + kh;
        v8bf b0 = *(const v8bf*)bp;
        v8bf b1 = *(const v8bf*)(bp + 16);
        v16bf b;
#pragma unroll
        for (int e = 0; e < 8; ++e) { b[e] = b0[e]; b[e + 8] = b1[e]; }
        acc = __builtin_amdgcn_wmma_f32_16x16x32_bf16(false, a, false, b,
                                                      (short)0, acc, false, false);
    }
    float bv = (n < 11) ? bout[n] : 0.f;
    int r0 = rbase + half * 8;
#pragma unroll
    for (int v = 0; v < 8; ++v) feats[(size_t)(r0 + v) * 16 + n] = acc[v] + bv;
}

// ---------------- Viterbi: one 32-thread block per batch row -----------------
__global__ void __launch_bounds__(32) viterbi_kernel(const float* __restrict__ feats,
                                                     const float* __restrict__ trans,
                                                     int* __restrict__ bp,
                                                     float* __restrict__ out) {
    const int b = blockIdx.x;
    const int i = threadIdx.x;
    __shared__ float fv[16];
    __shared__ float nxt[16];
    __shared__ float term[16];

    float tr[11];
    if (i < 11)
        for (int j = 0; j < 11; ++j) tr[j] = trans[i * 11 + j];
    if (i < 16) fv[i] = (i == 9) ? 0.f : NEGINF;   // START = 9
    __syncthreads();

    for (int t = 0; t < 512; ++t) {
        if (i < 11) {
            float mx = -3.0e38f; int arg = 0;
            for (int j = 0; j < 11; ++j) {
                float sc = fv[j] + tr[j];
                if (sc > mx) { mx = sc; arg = j; }   // first max (matches argmax)
            }
            bp[(t * 64 + b) * 11 + i] = arg;
            nxt[i] = mx + feats[((size_t)t * 64 + b) * 16 + i];
        }
        __syncthreads();
        if (i < 11) fv[i] = nxt[i];
        __syncthreads();
    }
    if (i < 11) term[i] = fv[i] + trans[10 * 11 + i];   // STOP = 10
    __syncthreads();
    if (i == 0) {
        float best = term[0]; int bidx = 0;
        for (int j = 1; j < 11; ++j)
            if (term[j] > best) { best = term[j]; bidx = j; }
        out[b] = best;
        int tag = bidx;
        for (int t = 511; t >= 0; --t) {
            out[64 + b * 512 + t] = (float)tag;
            tag = bp[(t * 64 + b) * 11 + tag];
        }
    }
}

// ---------------------------------------------------------------------------
extern "C" void kernel_launch(void* const* d_in, const int* in_sizes, int n_in,
                              void* d_out, int out_size, void* d_ws, size_t ws_size,
                              hipStream_t stream) {
    (void)in_sizes; (void)n_in; (void)out_size; (void)ws_size;
    const int*   x     = (const int*)d_in[0];
    const float* emb   = (const float*)d_in[1];
    const float* Wout  = (const float*)d_in[2];
    const float* bout  = (const float*)d_in[3];
    const float* trans = (const float*)d_in[4];
    // order: l0f, l0b, l1f, l1b ; each (Wi, Wh, bi, bh)
    const float* Wi[4] = {(const float*)d_in[5],  (const float*)d_in[9],
                          (const float*)d_in[13], (const float*)d_in[17]};
    const float* Wh[4] = {(const float*)d_in[6],  (const float*)d_in[10],
                          (const float*)d_in[14], (const float*)d_in[18]};
    const float* bi[4] = {(const float*)d_in[7],  (const float*)d_in[11],
                          (const float*)d_in[15], (const float*)d_in[19]};
    const float* bh[4] = {(const float*)d_in[8],  (const float*)d_in[12],
                          (const float*)d_in[16], (const float*)d_in[20]};

    // ---- workspace carve-out (256B aligned) ----
    char* wp = (char*)d_ws;
    auto carve = [&](size_t bytes) -> void* {
        void* p = wp; wp += (bytes + 255) & ~(size_t)255; return p;
    };
    __bf16* X0    = (__bf16*)carve((size_t)TB * KH * 2);     // embedded input, bf16
    __bf16* H1    = (__bf16*)carve((size_t)TB * K2H * 2);    // layer0 output
    __bf16* H2    = (__bf16*)carve((size_t)TB * K2H * 2);    // layer1 output
    __bf16* preF  = (__bf16*)carve((size_t)TB * G4H * 2);    // fwd pre-activations
    __bf16* preB  = (__bf16*)carve((size_t)TB * G4H * 2);    // bwd pre-activations
    __bf16* WiBf[4]; __bf16* WhBf[4]; float* biasS[4];
    int KpadIn[4] = {KH, KH, K2H, K2H};
    for (int q = 0; q < 4; ++q) {
        WiBf[q]  = (__bf16*)carve((size_t)G4H * KpadIn[q] * 2);
        WhBf[q]  = (__bf16*)carve((size_t)G4H * KH * 2);
        biasS[q] = (float*)carve((size_t)G4H * 4);
    }
    __bf16* WoutBf = (__bf16*)carve((size_t)16 * K2H * 2);
    float*  feats  = (float*)carve((size_t)TB * 16 * 4);
    int*    bps    = (int*)carve((size_t)512 * 64 * 11 * 4);
    float*  outF   = (float*)d_out;

    // ---- weight prep: convert/pad to bf16, sum biases ----
    for (int q = 0; q < 4; ++q) {
        int split = (q >= 2) ? 1 : 0;   // layer1 Wi columns: [hf|hb] -> padded halves
        int srcC  = (q >= 2) ? 600 : 300;
        int tot   = G4H * KpadIn[q];
        cvt_pad_kernel<<<(tot + 255) / 256, 256, 0, stream>>>(Wi[q], WiBf[q], G4H, srcC,
                                                              KpadIn[q], split, tot);
        int totH = G4H * KH;
        cvt_pad_kernel<<<(totH + 255) / 256, 256, 0, stream>>>(Wh[q], WhBf[q], G4H, 300,
                                                               KH, 0, totH);
        bias_sum_kernel<<<(G4H + 255) / 256, 256, 0, stream>>>(bi[q], bh[q], biasS[q], G4H);
    }
    {
        int tot = 16 * K2H;
        cvt_pad_kernel<<<(tot + 255) / 256, 256, 0, stream>>>(Wout, WoutBf, 11, 600,
                                                              K2H, 1, tot);
    }

    // ---- embedding gather (zero-pads K to 320) ----
    embed_kernel<<<((size_t)TB * 80 + 255) / 256, 256, 0, stream>>>(x, emb, X0);

    // ---- zero layer outputs so K-pad columns are 0 for the next GEMM ----
    {
        size_t n = (size_t)TB * K2H;
        zero_bf16_kernel<<<(unsigned)((n + 255) / 256), 256, 0, stream>>>(H1, n);
        zero_bf16_kernel<<<(unsigned)((n + 255) / 256), 256, 0, stream>>>(H2, n);
    }

    const size_t scanLds = (size_t)64 * HSTRIDE * 2 + (size_t)32 * GSTRIDE * 4; // 192 KB

    // ---- layer 0: input projection then recurrent scan (fwd+bwd concurrent) ----
    {
        dim3 grid(19200, 2);
        gemm_pre_kernel<<<grid, 256, 0, stream>>>(X0, WiBf[0], WiBf[1], biasS[0], biasS[1],
                                                  preF, preB, KH);
        lstm_scan_kernel<<<2, 1024, scanLds, stream>>>(WhBf[0], WhBf[1], preF, preB, H1);
    }
    // ---- layer 1 ----
    {
        dim3 grid(19200, 2);
        gemm_pre_kernel<<<grid, 256, 0, stream>>>(H1, WiBf[2], WiBf[3], biasS[2], biasS[3],
                                                  preF, preB, K2H);
        lstm_scan_kernel<<<2, 1024, scanLds, stream>>>(WhBf[2], WhBf[3], preF, preB, H2);
    }

    // ---- output projection + Viterbi ----
    gemm_feats_kernel<<<256, 256, 0, stream>>>(H2, WoutBf, bout, feats);
    viterbi_kernel<<<64, 32, 0, stream>>>(feats, trans, bps, outF);
}
